// ApsMultiheadAttention_73959336837560
// MI455X (gfx1250) — compile-verified, matrix operand order checked by reading, not run on hardware
//
#include <hip/hip_runtime.h>

// ============================================================================
// MI455X (gfx1250) multi-head attention forward, wave32 + WMMA f16 path.
//
// Sizes (compile-time): E=1024, H=16, D=64, L=S=2048, N=2.
// d_out = [ context (L,N,E) fp32 | att (N,L,S) fp32 ]
// Workspace layout (needs ~33 MB):
//   qf16  [4096][1024] f16   @ 0        (token = l*N+n)
//   kf16  [4096][1024] f16   @ 8  MB    (token = s*N+n)
//   vt16  [N][1024][2048]f16 @ 16 MB    (per-batch transposed V for P@V frags)
//   ctx16 [4096][1024] f16   @ 24 MB
//   Mst   [N*H*L] f32        @ 32 MB    (softmax row max)
//   Sst   [N*H*L] f32        @ 32MB+256KB (softmax row sum)
// ============================================================================

typedef __attribute__((ext_vector_type(16))) _Float16 v16h;
typedef __attribute__((ext_vector_type(8)))  _Float16 v8h;
typedef __attribute__((ext_vector_type(8)))  float    v8f;

#define EDIM 1024
#define LDIM 2048
#define SDIM 2048
#define NB   2
#define HDIM 16
#define DD   64

__device__ __forceinline__ v8f wmma_f16(v16h a, v16h b, v8f c) {
  return __builtin_amdgcn_wmma_f32_16x16x32_f16(
      /*neg_a=*/false, a, /*neg_b=*/false, b,
      /*c_mod=*/(short)0, c, /*reuse_a=*/false, /*reuse_b=*/false);
}

// Build a 16-half fragment from two contiguous 8-half (16B) runs.
// Matches ISA A/B 16-bit layout: lane holds K = base+[0,8) and base+16+[0,8),
// base = (lane<16 ? 0 : 8).
__device__ __forceinline__ v16h ldfrag(const _Float16* p0, const _Float16* p1) {
  v8h a = *(const v8h*)p0;
  v8h b = *(const v8h*)p1;
  v16h r;
#pragma unroll
  for (int i = 0; i < 8; ++i) { r[i] = a[i]; r[i + 8] = b[i]; }
  return r;
}

// ============================================================================
// Kernel 1: QKV projection.  C[4096][1024] = X @ W_z^T + b_z  (fp32 in, f16 out)
// grid (16, 64, 3), block 256 (8 waves). 64x64 C tile per block, BK=32.
// z==2 (V) stores transposed per batch: vt[n][e][s].
// ============================================================================
__global__ void gemm_qkv(const float* __restrict__ query,
                         const float* __restrict__ key,
                         const float* __restrict__ value,
                         const float* __restrict__ ipw,
                         const float* __restrict__ ipb,
                         _Float16* __restrict__ qf,
                         _Float16* __restrict__ kf,
                         _Float16* __restrict__ vt) {
  __shared__ alignas(16) _Float16 As[64][32];
  __shared__ alignas(16) _Float16 Bs[64][32];
  const int z  = blockIdx.z;
  const float* X = (z == 0) ? query : (z == 1) ? key : value;
  const float* W = ipw + (size_t)z * EDIM * EDIM;
  const float* Bv = ipb + z * EDIM;
  const int n0 = blockIdx.x * 64, m0 = blockIdx.y * 64;
  const int t = threadIdx.x, lane = t & 31, wv = t >> 5;
  const int wm = wv & 3, wn = wv >> 2;
  const int lo = lane & 15, kb = (lane >> 4) * 8;  // half-wave K/row offset
  const int lr = t >> 2, lc = (t & 3) * 8;         // staging row / col
  v8f c0 = {}, c1 = {};
  for (int k0 = 0; k0 < EDIM; k0 += 32) {
    const float* ap = X + (size_t)(m0 + lr) * EDIM + k0 + lc;
    const float* bp = W + (size_t)(n0 + lr) * EDIM + k0 + lc;
#pragma unroll
    for (int i = 0; i < 8; ++i) As[lr][lc + i] = (_Float16)ap[i];
#pragma unroll
    for (int i = 0; i < 8; ++i) Bs[lr][lc + i] = (_Float16)bp[i];
    __syncthreads();
    v16h a  = ldfrag(&As[wm * 16 + lo][kb],      &As[wm * 16 + lo][kb + 16]);
    v16h b0 = ldfrag(&Bs[wn * 32 + lo][kb],      &Bs[wn * 32 + lo][kb + 16]);
    v16h b1 = ldfrag(&Bs[wn * 32 + 16 + lo][kb], &Bs[wn * 32 + 16 + lo][kb + 16]);
    c0 = wmma_f16(a, b0, c0);
    c1 = wmma_f16(a, b1, c1);
    __syncthreads();
  }
#pragma unroll
  for (int r = 0; r < 8; ++r) {
    const int row  = m0 + wm * 16 + r + kb;       // C row: vgpr + (hi?8:0)
    const int col0 = n0 + wn * 32 + lo;
    const float v0 = c0[r] + Bv[col0];
    const float v1 = c1[r] + Bv[col0 + 16];
    if (z == 2) {  // transposed per-batch store for V
      const int nb = row & 1, s = row >> 1;
      vt[((size_t)nb * EDIM + col0)        * SDIM + s] = (_Float16)v0;
      vt[((size_t)nb * EDIM + col0 + 16)   * SDIM + s] = (_Float16)v1;
    } else {
      _Float16* O = (z == 0) ? qf : kf;
      O[(size_t)row * EDIM + col0]      = (_Float16)v0;
      O[(size_t)row * EDIM + col0 + 16] = (_Float16)v1;
    }
  }
}

// ============================================================================
// Kernel 2: flash attention forward per (n, h, 64 L-rows).
// grid (L/64=32, H=16, N=2), block 128 (4 waves; wave -> 16 L rows).
// Online softmax over S in 64-wide tiles; saves (row max, row sum) stats.
// ============================================================================
__global__ void flash_fwd(const _Float16* __restrict__ qf,
                          const _Float16* __restrict__ kf,
                          const _Float16* __restrict__ vt,
                          const float* __restrict__ amask,
                          const unsigned char* __restrict__ kpm,
                          _Float16* __restrict__ ctx,
                          float* __restrict__ Mst,
                          float* __restrict__ Sst) {
  __shared__ alignas(16) _Float16 P[4][16][64];  // per-wave P tile (f16)
  const int t = threadIdx.x, lane = t & 31, wv = t >> 5;
  const int lo = lane & 15, kb = (lane >> 4) * 8;
  const int h = blockIdx.y, n = blockIdx.z;
  const int lbase = blockIdx.x * 64 + wv * 16;
  const int eh = h * DD;
  const float scale = 0.125f;  // 1/sqrt(64)

  // Q A-fragments (K = d, two 32-wide chunks)
  const size_t qrow = (size_t)((lbase + lo) * NB + n) * EDIM + eh;
  const v16h aq0 = ldfrag(qf + qrow + kb,      qf + qrow + kb + 16);
  const v16h aq1 = ldfrag(qf + qrow + 32 + kb, qf + qrow + 32 + kb + 16);

  float m[8], ssum[8];
#pragma unroll
  for (int r = 0; r < 8; ++r) { m[r] = -__builtin_huge_valf(); ssum[r] = 0.f; }
  v8f ctxa[4] = {};

  for (int s0 = 0; s0 < SDIM; s0 += 64) {
    // ---- logits: 4 N-tiles x (2 K-steps of WMMA) ----
    v8f lg[4];
#pragma unroll
    for (int j = 0; j < 4; ++j) {
      const int scol = (s0 + j * 16 + lo) * NB + n;
      const _Float16* kp = kf + (size_t)scol * EDIM + eh;
      v16h b0 = ldfrag(kp + kb, kp + kb + 16);
      v16h b1 = ldfrag(kp + 32 + kb, kp + 32 + kb + 16);
      v8f acc = {};
      acc = wmma_f16(aq0, b0, acc);
      acc = wmma_f16(aq1, b1, acc);
      lg[j] = acc;
    }
    // ---- scale + attn_mask + key_padding_mask ----
#pragma unroll
    for (int j = 0; j < 4; ++j) {
      const int sel = s0 + j * 16 + lo;
      const float kadj = kpm[n * SDIM + sel] ? -1e30f : 0.f;
#pragma unroll
      for (int r = 0; r < 8; ++r) {
        const int lrow = lbase + r + kb;
        lg[j][r] = lg[j][r] * scale + amask[(size_t)lrow * SDIM + sel] + kadj;
      }
    }
    // ---- online softmax: row max (xor-shuffle within 16-lane half) ----
    float corr[8];
#pragma unroll
    for (int r = 0; r < 8; ++r) {
      float v = fmaxf(fmaxf(lg[0][r], lg[1][r]), fmaxf(lg[2][r], lg[3][r]));
      v = fmaxf(v, __shfl_xor(v, 1, 32));
      v = fmaxf(v, __shfl_xor(v, 2, 32));
      v = fmaxf(v, __shfl_xor(v, 4, 32));
      v = fmaxf(v, __shfl_xor(v, 8, 32));
      const float nm = fmaxf(m[r], v);
      corr[r] = __expf(m[r] - nm);
      m[r] = nm;
    }
#pragma unroll
    for (int j = 0; j < 4; ++j) {
#pragma unroll
      for (int r = 0; r < 8; ++r) lg[j][r] = __expf(lg[j][r] - m[r]);
    }
#pragma unroll
    for (int r = 0; r < 8; ++r) {
      float v = lg[0][r] + lg[1][r] + lg[2][r] + lg[3][r];
      v += __shfl_xor(v, 1, 32);
      v += __shfl_xor(v, 2, 32);
      v += __shfl_xor(v, 4, 32);
      v += __shfl_xor(v, 8, 32);
      ssum[r] = ssum[r] * corr[r] + v;
    }
#pragma unroll
    for (int j = 0; j < 4; ++j) {
#pragma unroll
      for (int r = 0; r < 8; ++r) ctxa[j][r] *= corr[r];
    }
    // ---- C-layout -> A-fragment layout via LDS (f16) ----
#pragma unroll
    for (int j = 0; j < 4; ++j) {
#pragma unroll
      for (int r = 0; r < 8; ++r) P[wv][r + kb][j * 16 + lo] = (_Float16)lg[j][r];
    }
    __syncthreads();
    v16h ap0 = ldfrag(&P[wv][lo][kb],      &P[wv][lo][kb + 16]);
    v16h ap1 = ldfrag(&P[wv][lo][32 + kb], &P[wv][lo][32 + kb + 16]);
    // ---- context += P @ V ----
#pragma unroll
    for (int j = 0; j < 4; ++j) {
      const int e = eh + j * 16 + lo;
      const _Float16* vp = vt + ((size_t)n * EDIM + e) * SDIM + s0;
      v16h bv0 = ldfrag(vp + kb, vp + kb + 16);
      v16h bv1 = ldfrag(vp + 32 + kb, vp + 32 + kb + 16);
      ctxa[j] = wmma_f16(ap0, bv0, ctxa[j]);
      ctxa[j] = wmma_f16(ap1, bv1, ctxa[j]);
    }
    __syncthreads();
  }
  // ---- normalize + store context (f16 to ws) ----
  float inv[8];
#pragma unroll
  for (int r = 0; r < 8; ++r) inv[r] = 1.f / ssum[r];
#pragma unroll
  for (int j = 0; j < 4; ++j) {
#pragma unroll
    for (int r = 0; r < 8; ++r) {
      const int lrow = lbase + r + kb;
      const int e = eh + j * 16 + lo;
      ctx[(size_t)(lrow * NB + n) * EDIM + e] = (_Float16)(ctxa[j][r] * inv[r]);
    }
  }
  // ---- store softmax stats (one lane per half-wave) ----
  if (lo == 0) {
    const int sb = (n * HDIM + h) * LDIM + lbase + kb;
#pragma unroll
    for (int r = 0; r < 8; ++r) { Mst[sb + r] = m[r]; Sst[sb + r] = ssum[r]; }
  }
}

// ============================================================================
// Kernel 3: att[n][l][s] = (1/H) * sum_h exp(logit - M)/S  (deterministic,
// WMMA logit recompute, no atomics). grid (S/64, L/64, N), block 128.
// ============================================================================
__global__ void att_pass2(const _Float16* __restrict__ qf,
                          const _Float16* __restrict__ kf,
                          const float* __restrict__ amask,
                          const unsigned char* __restrict__ kpm,
                          const float* __restrict__ Mst,
                          const float* __restrict__ Sst,
                          float* __restrict__ att) {
  const int t = threadIdx.x, lane = t & 31, wv = t >> 5;
  const int lo = lane & 15, kb = (lane >> 4) * 8;
  const int n = blockIdx.z;
  const int lbase = blockIdx.y * 64 + wv * 16;
  const int s0 = blockIdx.x * 64;
  // mask (head-independent) precomputed once
  float mv[4][8];
#pragma unroll
  for (int j = 0; j < 4; ++j) {
    const int sel = s0 + j * 16 + lo;
    const float kadj = kpm[n * SDIM + sel] ? -1e30f : 0.f;
#pragma unroll
    for (int r = 0; r < 8; ++r)
      mv[j][r] = amask[(size_t)(lbase + r + kb) * SDIM + sel] + kadj;
  }
  v8f acc[4] = {};
  for (int h = 0; h < HDIM; ++h) {
    const int eh = h * DD;
    const size_t qrow = (size_t)((lbase + lo) * NB + n) * EDIM + eh;
    v16h aq0 = ldfrag(qf + qrow + kb,      qf + qrow + kb + 16);
    v16h aq1 = ldfrag(qf + qrow + 32 + kb, qf + qrow + 32 + kb + 16);
    float mh[8], rsh[8];
    const int sb = (n * HDIM + h) * LDIM + lbase + kb;
#pragma unroll
    for (int r = 0; r < 8; ++r) { mh[r] = Mst[sb + r]; rsh[r] = 1.f / Sst[sb + r]; }
#pragma unroll
    for (int j = 0; j < 4; ++j) {
      const int scol = (s0 + j * 16 + lo) * NB + n;
      const _Float16* kp = kf + (size_t)scol * EDIM + eh;
      v16h b0 = ldfrag(kp + kb, kp + kb + 16);
      v16h b1 = ldfrag(kp + 32 + kb, kp + 32 + kb + 16);
      v8f lacc = {};
      lacc = wmma_f16(aq0, b0, lacc);
      lacc = wmma_f16(aq1, b1, lacc);
#pragma unroll
      for (int r = 0; r < 8; ++r)
        acc[j][r] += __expf(lacc[r] * 0.125f + mv[j][r] - mh[r]) * rsh[r];
    }
  }
#pragma unroll
  for (int j = 0; j < 4; ++j) {
#pragma unroll
    for (int r = 0; r < 8; ++r)
      att[((size_t)n * LDIM + lbase + r + kb) * SDIM + s0 + j * 16 + lo] =
          acc[j][r] * (1.f / 16.f);
  }
}

// ============================================================================
// Kernel 4: output projection. out[4096][1024] = ctx_f16 @ out_w^T + out_b (fp32)
// grid (16, 64), block 256.
// ============================================================================
__global__ void gemm_out(const _Float16* __restrict__ ctx,
                         const float* __restrict__ W,
                         const float* __restrict__ bias,
                         float* __restrict__ out) {
  __shared__ alignas(16) _Float16 As[64][32];
  __shared__ alignas(16) _Float16 Bs[64][32];
  const int n0 = blockIdx.x * 64, m0 = blockIdx.y * 64;
  const int t = threadIdx.x, lane = t & 31, wv = t >> 5;
  const int wm = wv & 3, wn = wv >> 2;
  const int lo = lane & 15, kb = (lane >> 4) * 8;
  const int lr = t >> 2, lc = (t & 3) * 8;
  v8f c0 = {}, c1 = {};
  for (int k0 = 0; k0 < EDIM; k0 += 32) {
    const _Float16* ap = ctx + (size_t)(m0 + lr) * EDIM + k0 + lc;
    const float* bp = W + (size_t)(n0 + lr) * EDIM + k0 + lc;
    *(v8h*)&As[lr][lc] = *(const v8h*)ap;
#pragma unroll
    for (int i = 0; i < 8; ++i) Bs[lr][lc + i] = (_Float16)bp[i];
    __syncthreads();
    v16h a  = ldfrag(&As[wm * 16 + lo][kb],      &As[wm * 16 + lo][kb + 16]);
    v16h b0 = ldfrag(&Bs[wn * 32 + lo][kb],      &Bs[wn * 32 + lo][kb + 16]);
    v16h b1 = ldfrag(&Bs[wn * 32 + 16 + lo][kb], &Bs[wn * 32 + 16 + lo][kb + 16]);
    c0 = wmma_f16(a, b0, c0);
    c1 = wmma_f16(a, b1, c1);
    __syncthreads();
  }
#pragma unroll
  for (int r = 0; r < 8; ++r) {
    const int row = m0 + wm * 16 + r + kb;
    const int col = n0 + wn * 32 + lo;
    out[(size_t)row * EDIM + col]      = c0[r] + bias[col];
    out[(size_t)row * EDIM + col + 16] = c1[r] + bias[col + 16];
  }
}

// ============================================================================
extern "C" void kernel_launch(void* const* d_in, const int* in_sizes, int n_in,
                              void* d_out, int out_size, void* d_ws, size_t ws_size,
                              hipStream_t stream) {
  (void)in_sizes; (void)n_in; (void)out_size; (void)ws_size;
  const float* query = (const float*)d_in[0];
  const float* key   = (const float*)d_in[1];
  const float* value = (const float*)d_in[2];
  const unsigned char* kpm = (const unsigned char*)d_in[3];
  const float* amask = (const float*)d_in[4];
  const float* ipw   = (const float*)d_in[5];
  const float* ipb   = (const float*)d_in[6];
  const float* outw  = (const float*)d_in[7];
  const float* outb  = (const float*)d_in[8];

  float* out_ctx = (float*)d_out;
  float* out_att = out_ctx + (size_t)LDIM * NB * EDIM;

  char* ws = (char*)d_ws;
  _Float16* qf  = (_Float16*)(ws);
  _Float16* kf  = (_Float16*)(ws + (size_t)(8u << 20));
  _Float16* vt  = (_Float16*)(ws + (size_t)(16u << 20));
  _Float16* cf  = (_Float16*)(ws + (size_t)(24u << 20));
  float*    Mst = (float*)   (ws + (size_t)(32u << 20));
  float*    Sst = Mst + NB * HDIM * LDIM;

  gemm_qkv<<<dim3(16, 64, 3), dim3(256), 0, stream>>>(
      query, key, value, ipw, ipb, qf, kf, vt);
  flash_fwd<<<dim3(LDIM / 64, HDIM, NB), dim3(128), 0, stream>>>(
      qf, kf, vt, amask, kpm, cf, Mst, Sst);
  att_pass2<<<dim3(SDIM / 64, LDIM / 64, NB), dim3(128), 0, stream>>>(
      qf, kf, amask, kpm, Mst, Sst, out_att);
  gemm_out<<<dim3(16, 64, 1), dim3(256), 0, stream>>>(
      cf, outw, outb, out_ctx);
}
